// MultiHeadAttention_22514218566347
// MI455X (gfx1250) — matrix operand order repaired and checked by reading, not compiled, not run on hardware
//
#include <hip/hip_runtime.h>

// ---------------- constants -------------------------------------------------
constexpr int BB   = 2;
constexpr int SS   = 2048;
constexpr int DM   = 1024;
constexpr int NH   = 16;
constexpr int DK   = 64;
constexpr int MR   = BB * SS;        // 4096 rows for projections
constexpr float SCALE = 0.125f;      // 1/sqrt(64)

typedef __attribute__((ext_vector_type(8)))  __bf16 v8bf;
typedef __attribute__((ext_vector_type(16))) __bf16 v16bf;
typedef __attribute__((ext_vector_type(8)))  float  v8f;

// ---------------- fragment loaders (CDNA5 WMMA VGPR layouts) ----------------
// A fragment 16(M) x 32(K), source row-major [M][>=32] with leading dim ld.
// lane<16 : row=lane,    K = {0..7, 16..23}
// lane>=16: row=lane-16, K = {8..15, 24..31}
__device__ inline v16bf load_a_frag(const __bf16* __restrict__ p, int ld) {
  int lane = threadIdx.x & 31;
  int row  = lane & 15;
  int koff = (lane >> 4) * 8;
  const __bf16* q = p + (size_t)row * ld + koff;
  v8bf lo = *(const v8bf*)(q);
  v8bf hi = *(const v8bf*)(q + 16);
  return __builtin_shufflevector(lo, hi, 0,1,2,3,4,5,6,7,8,9,10,11,12,13,14,15);
}

// B fragment 32(K) x 16(N), source stored as [N][K] row-major (K contiguous).
// lane<16 : col=lane,    K = {0..7, 16..23} ; lane>=16: col=lane-16, K = {8..15, 24..31}
__device__ inline v16bf load_b_frag(const __bf16* __restrict__ p, int ld) {
  int lane = threadIdx.x & 31;
  int n    = lane & 15;
  int koff = (lane >> 4) * 8;
  const __bf16* q = p + (size_t)n * ld + koff;
  v8bf lo = *(const v8bf*)(q);
  v8bf hi = *(const v8bf*)(q + 16);
  return __builtin_shufflevector(lo, hi, 0,1,2,3,4,5,6,7,8,9,10,11,12,13,14,15);
}

__device__ inline v8f wmma_bf16(v16bf a, v16bf b, v8f c) {
  return __builtin_amdgcn_wmma_f32_16x16x32_bf16(false, a, false, b, (short)0, c,
                                                 false, false);
}

// ---------------- fp32 -> bf16 cast ----------------------------------------
__global__ void cast_f32_bf16(const float* __restrict__ in,
                              __bf16* __restrict__ out, int n4) {
  int i = blockIdx.x * blockDim.x + threadIdx.x;
  if (i < n4) {
    float4 v = ((const float4*)in)[i];
    out[i * 4 + 0] = (__bf16)v.x;
    out[i * 4 + 1] = (__bf16)v.y;
    out[i * 4 + 2] = (__bf16)v.z;
    out[i * 4 + 3] = (__bf16)v.w;
  }
}

// ---------------- WMMA GEMM: Y = X[M,K] * W[N,K]^T + bias -------------------
// One wave -> 32x64 output tile, register-double-buffered operand fetch.
// out_mode: 0 = bf16 scatter to [B,H,S,DK] (Q/K), 1 = bf16 scatter to
// [B,H,DK,S] (V transposed), 2 = f32 row-major.
__global__ void wmma_gemm_xwT(const __bf16* __restrict__ X,
                              const __bf16* __restrict__ W,
                              const float* __restrict__ bias,
                              void* __restrict__ outp,
                              int Kdim, int Ncols, int out_mode) {
  const int mt   = blockIdx.x * 32;
  const int nt   = blockIdx.y * 64;
  const int lane = threadIdx.x & 31;

  const __bf16* Ap = X + (size_t)mt * Kdim;
  const __bf16* Wp = W + (size_t)nt * Kdim;

  v8f c[8] = {};

  // prologue: fragments for k = 0
  v16bf a0 = load_a_frag(Ap, Kdim);
  v16bf a1 = load_a_frag(Ap + (size_t)16 * Kdim, Kdim);
  v16bf b0 = load_b_frag(Wp, Kdim);
  v16bf b1 = load_b_frag(Wp + (size_t)16 * Kdim, Kdim);
  v16bf b2 = load_b_frag(Wp + (size_t)32 * Kdim, Kdim);
  v16bf b3 = load_b_frag(Wp + (size_t)48 * Kdim, Kdim);

  for (int k = 32; k < Kdim; k += 32) {
    // issue next iteration's loads before this iteration's WMMAs so the
    // s_wait before each WMMA covers loads from a full iteration earlier
    v16bf a0n = load_a_frag(Ap + k, Kdim);
    v16bf a1n = load_a_frag(Ap + (size_t)16 * Kdim + k, Kdim);
    v16bf b0n = load_b_frag(Wp + k, Kdim);
    v16bf b1n = load_b_frag(Wp + (size_t)16 * Kdim + k, Kdim);
    v16bf b2n = load_b_frag(Wp + (size_t)32 * Kdim + k, Kdim);
    v16bf b3n = load_b_frag(Wp + (size_t)48 * Kdim + k, Kdim);

    c[0] = wmma_bf16(a0, b0, c[0]);
    c[1] = wmma_bf16(a0, b1, c[1]);
    c[2] = wmma_bf16(a0, b2, c[2]);
    c[3] = wmma_bf16(a0, b3, c[3]);
    c[4] = wmma_bf16(a1, b0, c[4]);
    c[5] = wmma_bf16(a1, b1, c[5]);
    c[6] = wmma_bf16(a1, b2, c[6]);
    c[7] = wmma_bf16(a1, b3, c[7]);

    a0 = a0n; a1 = a1n; b0 = b0n; b1 = b1n; b2 = b2n; b3 = b3n;
  }
  // epilogue K-step
  c[0] = wmma_bf16(a0, b0, c[0]);
  c[1] = wmma_bf16(a0, b1, c[1]);
  c[2] = wmma_bf16(a0, b2, c[2]);
  c[3] = wmma_bf16(a0, b3, c[3]);
  c[4] = wmma_bf16(a1, b0, c[4]);
  c[5] = wmma_bf16(a1, b1, c[5]);
  c[6] = wmma_bf16(a1, b2, c[6]);
  c[7] = wmma_bf16(a1, b3, c[7]);

  const int nl    = lane & 15;
  const int mbase = 8 * (lane >> 4);
#pragma unroll
  for (int i = 0; i < 8; ++i) {
    int aIdx = i >> 2;        // which 16-row band
    int j    = i & 3;         // which 16-col band
#pragma unroll
    for (int r = 0; r < 8; ++r) {
      int m = mt + 16 * aIdx + mbase + r;
      int n = nt + 16 * j + nl;
      float val = c[i][r] + bias[n];
      if (out_mode == 2) {
        ((float*)outp)[(size_t)m * Ncols + n] = val;
      } else {
        int b = m >> 11;       // m / S
        int s = m & (SS - 1);  // m % S
        int h = n >> 6;        // n / DK
        int d = n & 63;        // n % DK
        if (out_mode == 0)
          ((__bf16*)outp)[(((size_t)b * NH + h) * SS + s) * DK + d] = (__bf16)val;
        else
          ((__bf16*)outp)[(((size_t)b * NH + h) * DK + d) * SS + s] = (__bf16)val;
      }
    }
  }
}

// ---------------- flash attention core --------------------------------------
// grid: (S/16, B*H), 32 threads. Q,K: [B,H,S,DK] bf16. Vt: [B,H,DK,S] bf16.
// Writes concat [B,S,DM] bf16.
__global__ void flash_attn(const __bf16* __restrict__ Q,
                           const __bf16* __restrict__ Kk,
                           const __bf16* __restrict__ Vt,
                           const int* __restrict__ mask,
                           __bf16* __restrict__ Cc) {
  __shared__ __bf16 pTile[16 * 32];

  const int qt   = blockIdx.x * 16;
  const int bh   = blockIdx.y;
  const int b    = bh >> 4;
  const int h    = bh & 15;
  const int lane = threadIdx.x & 31;
  const int nl    = lane & 15;
  const int mbase = 8 * (lane >> 4);

  const __bf16* Qh = Q  + (size_t)bh * SS * DK;
  const __bf16* Kh = Kk + (size_t)bh * SS * DK;
  const __bf16* Vh = Vt + (size_t)bh * DK * SS;
  const int*    mb = mask + (size_t)b * SS * SS;

  // Q tile fragments (d = 0..31 and 32..63)
  v16bf qa0 = load_a_frag(Qh + (size_t)qt * DK + 0,  DK);
  v16bf qa1 = load_a_frag(Qh + (size_t)qt * DK + 32, DK);

  v8f acc[4] = {};
  float mx[8], ls[8];
#pragma unroll
  for (int r = 0; r < 8; ++r) { mx[r] = -__builtin_inff(); ls[r] = 0.f; }

  for (int kk = 0; kk < SS; kk += 32) {
    // prefetch next K tile rows into cache while this tile computes
    __builtin_prefetch(Kh + (size_t)(kk + 32) * DK, 0, 1);
    __builtin_prefetch(Kh + (size_t)(kk + 48) * DK, 0, 1);

    // scores: Q(16x64) @ K_tile(32x64)^T -> two 16x16 f32 fragments
    v8f sc[2] = {};
#pragma unroll
    for (int t = 0; t < 2; ++t) {
      v16bf kb0 = load_b_frag(Kh + (size_t)(kk + 16 * t) * DK + 0,  DK);
      v16bf kb1 = load_b_frag(Kh + (size_t)(kk + 16 * t) * DK + 32, DK);
      sc[t] = wmma_bf16(qa0, kb0, sc[t]);
      sc[t] = wmma_bf16(qa1, kb1, sc[t]);
    }

    // issue V fragment loads now; they overlap the softmax VALU work below
    v16bf vb0 = load_b_frag(Vh + (size_t)0  * SS + kk, SS);
    v16bf vb1 = load_b_frag(Vh + (size_t)16 * SS + kk, SS);
    v16bf vb2 = load_b_frag(Vh + (size_t)32 * SS + kk, SS);
    v16bf vb3 = load_b_frag(Vh + (size_t)48 * SS + kk, SS);

    // scale + mask
    float p0[8], p1[8], rmax[8];
#pragma unroll
    for (int r = 0; r < 8; ++r) {
      int qrow = qt + mbase + r;
      float s0 = sc[0][r] * SCALE;
      float s1 = sc[1][r] * SCALE;
      if (mb[(size_t)qrow * SS + kk + nl] == 0)      s0 = -1e9f;
      if (mb[(size_t)qrow * SS + kk + 16 + nl] == 0) s1 = -1e9f;
      p0[r] = s0; p1[r] = s1;
      rmax[r] = fmaxf(s0, s1);
    }
    // row-max across the 16 lanes owning this row (xor<16 stays in group)
#pragma unroll
    for (int r = 0; r < 8; ++r)
      for (int off = 1; off < 16; off <<= 1)
        rmax[r] = fmaxf(rmax[r], __shfl_xor(rmax[r], off, 32));

    float rsum[8];
#pragma unroll
    for (int r = 0; r < 8; ++r) {
      float mnew = fmaxf(mx[r], rmax[r]);
      float f = __expf(mx[r] - mnew);
      p0[r] = __expf(p0[r] - mnew);
      p1[r] = __expf(p1[r] - mnew);
      rsum[r] = p0[r] + p1[r];
      mx[r] = mnew;
      ls[r] *= f;
#pragma unroll
      for (int j = 0; j < 4; ++j) acc[j][r] *= f;
    }
#pragma unroll
    for (int r = 0; r < 8; ++r)
      for (int off = 1; off < 16; off <<= 1)
        rsum[r] += __shfl_xor(rsum[r], off, 32);
#pragma unroll
    for (int r = 0; r < 8; ++r) ls[r] += rsum[r];

    // C-layout -> A-fragment layout relayout of P through LDS (bf16)
    __syncthreads();
#pragma unroll
    for (int r = 0; r < 8; ++r) {
      int ml = mbase + r;
      pTile[ml * 32 + nl]      = (__bf16)p0[r];
      pTile[ml * 32 + 16 + nl] = (__bf16)p1[r];
    }
    __syncthreads();
    v16bf pa;
    {
      int row  = lane & 15;
      int koff = (lane >> 4) * 8;
      const __bf16* q = &pTile[row * 32 + koff];
      v8bf lo = *(const v8bf*)(q);
      v8bf hi = *(const v8bf*)(q + 16);
      pa = __builtin_shufflevector(lo, hi, 0,1,2,3,4,5,6,7,8,9,10,11,12,13,14,15);
    }

    // acc += P(16x32) @ V(32x64); Vt gives contiguous keys per d-row
    acc[0] = wmma_bf16(pa, vb0, acc[0]);
    acc[1] = wmma_bf16(pa, vb1, acc[1]);
    acc[2] = wmma_bf16(pa, vb2, acc[2]);
    acc[3] = wmma_bf16(pa, vb3, acc[3]);
    __syncthreads();
  }

  // normalize and scatter to concat [B,S,DM]
#pragma unroll
  for (int j = 0; j < 4; ++j) {
#pragma unroll
    for (int r = 0; r < 8; ++r) {
      int qrow = qt + mbase + r;
      int d    = 16 * j + nl;
      float val = acc[j][r] / ls[r];
      Cc[((size_t)b * SS + qrow) * DM + h * DK + d] = (__bf16)val;
    }
  }
}

// ---------------- launch ----------------------------------------------------
extern "C" void kernel_launch(void* const* d_in, const int* in_sizes, int n_in,
                              void* d_out, int out_size, void* d_ws, size_t ws_size,
                              hipStream_t stream) {
  const float* xq   = (const float*)d_in[0];
  const float* xk   = (const float*)d_in[1];
  const float* xv   = (const float*)d_in[2];
  const int*   mask = (const int*)d_in[3];
  const float* Wq   = (const float*)d_in[4];
  const float* bq   = (const float*)d_in[5];
  const float* Wk   = (const float*)d_in[6];
  const float* bk   = (const float*)d_in[7];
  const float* Wv   = (const float*)d_in[8];
  const float* bv   = (const float*)d_in[9];
  const float* Wo   = (const float*)d_in[10];
  const float* bo   = (const float*)d_in[11];

  const size_t MD = (size_t)MR * DM;   // 4M elems
  const size_t DD = (size_t)DM * DM;   // 1M elems

  char* ws = (char*)d_ws;
  __bf16* Xq16 = (__bf16*)ws; ws += MD * 2;
  __bf16* Xk16 = (__bf16*)ws; ws += MD * 2;
  __bf16* Xv16 = (__bf16*)ws; ws += MD * 2;
  __bf16* Wq16 = (__bf16*)ws; ws += DD * 2;
  __bf16* Wk16 = (__bf16*)ws; ws += DD * 2;
  __bf16* Wv16 = (__bf16*)ws; ws += DD * 2;
  __bf16* Wo16 = (__bf16*)ws; ws += DD * 2;
  __bf16* Q16  = (__bf16*)ws; ws += MD * 2;
  __bf16* K16  = (__bf16*)ws; ws += MD * 2;
  __bf16* Vt16 = (__bf16*)ws; ws += MD * 2;
  __bf16* C16  = (__bf16*)ws; ws += MD * 2;

  const int castThreads = 256;
  int nbM = (int)(MD / 4 + castThreads - 1) / castThreads;
  int nbD = (int)(DD / 4 + castThreads - 1) / castThreads;
  cast_f32_bf16<<<nbM, castThreads, 0, stream>>>(xq, Xq16, (int)(MD / 4));
  cast_f32_bf16<<<nbM, castThreads, 0, stream>>>(xk, Xk16, (int)(MD / 4));
  cast_f32_bf16<<<nbM, castThreads, 0, stream>>>(xv, Xv16, (int)(MD / 4));
  cast_f32_bf16<<<nbD, castThreads, 0, stream>>>(Wq, Wq16, (int)(DD / 4));
  cast_f32_bf16<<<nbD, castThreads, 0, stream>>>(Wk, Wk16, (int)(DD / 4));
  cast_f32_bf16<<<nbD, castThreads, 0, stream>>>(Wv, Wv16, (int)(DD / 4));
  cast_f32_bf16<<<nbD, castThreads, 0, stream>>>(Wo, Wo16, (int)(DD / 4));

  dim3 gemmGrid(MR / 32, DM / 64);  // 128 x 16 waves, 32x64 tile each
  wmma_gemm_xwT<<<gemmGrid, 32, 0, stream>>>(Xq16, Wq16, bq, Q16,  DM, DM, 0);
  wmma_gemm_xwT<<<gemmGrid, 32, 0, stream>>>(Xk16, Wk16, bk, K16,  DM, DM, 0);
  wmma_gemm_xwT<<<gemmGrid, 32, 0, stream>>>(Xv16, Wv16, bv, Vt16, DM, DM, 1);

  dim3 faGrid(SS / 16, BB * NH);    // 128 x 32 waves
  flash_attn<<<faGrid, 32, 0, stream>>>(Q16, K16, Vt16, mask, C16);

  wmma_gemm_xwT<<<gemmGrid, 32, 0, stream>>>(C16, Wo16, bo, d_out, DM, DM, 2);
}
